// GINCombinedv2_13262859010608
// MI455X (gfx1250) — compile-verified, matrix operand order checked by reading, not run on hardware
//
#include <hip/hip_runtime.h>
#include <hip/hip_bf16.h>

typedef __attribute__((ext_vector_type(16))) __bf16 v16bf;
typedef __attribute__((ext_vector_type(8)))  float  v8f;

// ---------- scalar helpers ----------
__device__ __forceinline__ unsigned short f32_to_bf16_rne(float f) {
    unsigned int u = __builtin_bit_cast(unsigned int, f);
    unsigned int r = u + 0x7FFFu + ((u >> 16) & 1u);
    return (unsigned short)(r >> 16);
}
// order-preserving float<->uint encoding for atomic max on floats
__device__ __forceinline__ unsigned enc_f(float f) {
    unsigned u = __builtin_bit_cast(unsigned, f);
    return (u & 0x80000000u) ? ~u : (u | 0x80000000u);
}
__device__ __forceinline__ float dec_f(unsigned u) {
    unsigned v = (u & 0x80000000u) ? (u & 0x7FFFFFFFu) : ~u;
    return __builtin_bit_cast(float, v);
}

// ---------- weight conversion + fragment swizzle ----------
// dst layout: [K/32][128 cols][2 halves][16 elems] -> each lane's B fragment
// is 32 contiguous bytes in LDS.
template <int K>
__global__ void cvt_swizzle_W_kernel(const float* __restrict__ src, // [K,128]
                                     unsigned short* __restrict__ dst) {
    int idx = blockIdx.x * blockDim.x + threadIdx.x;
    if (idx < K * 128) {
        int i    = idx & 15;
        int half = (idx >> 4) & 1;
        int col  = (idx >> 5) & 127;
        int kk32 = idx >> 12;
        int k = kk32 * 32 + half * 8 + (i & 7) + ((i >> 3) << 4);
        dst[idx] = f32_to_bf16_rne(src[k * 128 + col]);
    }
}

__global__ void copy_f32_kernel(const float* __restrict__ src,
                                float* __restrict__ dst, int n) {
    int i = blockIdx.x * blockDim.x + threadIdx.x;
    if (i < n) dst[i] = src[i];
}

// scatter-add over edges: acc[dst*F+f] += feat[src*F+f]
template <int F>
__global__ void scatter_add_kernel(const float* __restrict__ feat,
                                   const int* __restrict__ ei, int E,
                                   float* __restrict__ acc) {
    constexpr int SH = (F == 64) ? 6 : 7;
    int idx = blockIdx.x * blockDim.x + threadIdx.x;
    int total = E << SH;
    if (idx < total) {
        int e = idx >> SH;
        int f = idx & (F - 1);
        int s = ei[e];
        int d = ei[E + e];
        atomicAdd(&acc[(size_t)d * F + f], feat[(size_t)s * F + f]);
    }
}

// ---------- fused WMMA GEMM:  Out = epilogue(A[N,K] * W[K,128] + bias) ----------
// 32 rows per block (two 16x16 M-tiles per wave sharing one B fragment).
// EPI 0: relu(.)            -> bf16 out (LDS-staged, coalesced uint4 stores)
// EPI 1: relu(LayerNorm(.)) -> f32 out  (LN over the 128-wide row, fused in LDS)
// Epilogue LDS aliases the W/A staging buffers (separated by barriers).
template <int K, bool SRC_BF16, int EPI>
__launch_bounds__(256)
__global__ void gemm32_wmma_kernel(const void* __restrict__ Avoid,
                                   const unsigned short* __restrict__ Wsw, // swizzled bf16
                                   const float* __restrict__ bias,         // [128]
                                   const float* __restrict__ ln_g,
                                   const float* __restrict__ ln_b,
                                   void* __restrict__ Out, int Nrows) {
    constexpr size_t SM_STAGE = (size_t)(K * 128 + 32 * K) * 2;  // sW + sA (bf16)
    constexpr size_t SM_EPI   = (size_t)32 * 128 * 4;            // sC (f32) / sO (bf16)
    constexpr size_t SM_BYTES = SM_STAGE > SM_EPI ? SM_STAGE : SM_EPI;
    __shared__ __align__(16) char smem[SM_BYTES];
    unsigned short* sW = (unsigned short*)smem;          // [K*128] swizzled
    unsigned short* sA = sW + K * 128;                   // [32*K] row-major
    float*          sC = (float*)smem;                   // aliased (post-barrier)
    unsigned short* sO = (unsigned short*)smem;          // aliased (post-barrier)

    const int tid  = threadIdx.x;
    const int row0 = blockIdx.x * 32;
    const bool full = (row0 + 32 <= Nrows);

    // stage W (already fragment-ordered): contiguous 16B copies
    for (int i = tid; i < K * 16; i += 256)
        ((uint4*)sW)[i] = ((const uint4*)Wsw)[i];

    // stage 32 rows of A (bf16 copy, or f32 load + packed bf16 convert)
    if (SRC_BF16) {
        const unsigned short* A = (const unsigned short*)Avoid;
        if (full) {
            for (int i = tid; i < 32 * (K / 8); i += 256) {
                int r = i / (K / 8), c8 = i % (K / 8);
                ((uint4*)(sA + r * K))[c8] =
                    ((const uint4*)(A + (size_t)(row0 + r) * K))[c8];
            }
        } else {
            for (int i = tid; i < 32 * (K / 8); i += 256) {
                int r = i / (K / 8), c8 = i % (K / 8);
                uint4 v = make_uint4(0u, 0u, 0u, 0u);
                if (row0 + r < Nrows) v = ((const uint4*)(A + (size_t)(row0 + r) * K))[c8];
                ((uint4*)(sA + r * K))[c8] = v;
            }
        }
    } else {
        const float* A = (const float*)Avoid;
        for (int i = tid; i < 32 * (K / 4); i += 256) {
            int r = i / (K / 4), c4 = i % (K / 4);
            float4 v = make_float4(0.f, 0.f, 0.f, 0.f);
            if (full || row0 + r < Nrows)
                v = ((const float4*)(A + (size_t)(row0 + r) * K))[c4];
            unsigned p0 = (unsigned)f32_to_bf16_rne(v.x) | ((unsigned)f32_to_bf16_rne(v.y) << 16);
            unsigned p1 = (unsigned)f32_to_bf16_rne(v.z) | ((unsigned)f32_to_bf16_rne(v.w) << 16);
            ((uint2*)(sA + r * K))[c4] = make_uint2(p0, p1);
        }
    }
    __syncthreads();

    const int wave = tid >> 5;   // 0..7 -> 16-col tile
    const int lane = tid & 31;
    const int half = lane >> 4;
    const int l16  = lane & 15;
    const int col  = wave * 16 + l16;

    v8f c0 = {0.f, 0.f, 0.f, 0.f, 0.f, 0.f, 0.f, 0.f};
    v8f c1 = {0.f, 0.f, 0.f, 0.f, 0.f, 0.f, 0.f, 0.f};
    union Frag { v16bf v; uint4 q[2]; };
#pragma unroll
    for (int kk32 = 0; kk32 < K / 32; ++kk32) {
        const int kk = kk32 * 32;
        Frag a0, a1, b;
        // A fragment: lane holds row l16; K map {h*8..h*8+7, 16+h*8..16+h*8+7}
        const unsigned short* pa0 = sA + l16 * K + kk + half * 8;
        const unsigned short* pa1 = sA + (l16 + 16) * K + kk + half * 8;
        a0.q[0] = *(const uint4*)pa0;
        a0.q[1] = *(const uint4*)(pa0 + 16);
        a1.q[0] = *(const uint4*)pa1;
        a1.q[1] = *(const uint4*)(pa1 + 16);
        // B fragment: 32 contiguous bytes (pre-swizzled)
        const unsigned short* pb = sW + (((kk32 * 128 + col) * 2 + half) << 4);
        b.q[0] = *(const uint4*)pb;
        b.q[1] = *(const uint4*)(pb + 8);
        c0 = __builtin_amdgcn_wmma_f32_16x16x32_bf16(false, a0.v, false, b.v,
                                                     (short)0, c0, false, false);
        c1 = __builtin_amdgcn_wmma_f32_16x16x32_bf16(false, a1.v, false, b.v,
                                                     (short)0, c1, false, false);
    }
    __syncthreads();   // staging buffers dead; epilogue LDS aliases them

    const float bcol = bias[col];
    if (EPI == 0) {
        // relu + bf16, staged in LDS, then coalesced wide stores
#pragma unroll
        for (int r = 0; r < 8; ++r) {
            int m = r + half * 8;          // C layout: lanes 16-31 hold rows m+8
            float v0 = c0[r] + bcol;  v0 = v0 > 0.f ? v0 : 0.f;
            float v1 = c1[r] + bcol;  v1 = v1 > 0.f ? v1 : 0.f;
            sO[m * 128 + col]        = f32_to_bf16_rne(v0);
            sO[(m + 16) * 128 + col] = f32_to_bf16_rne(v1);
        }
        __syncthreads();
        unsigned short* Ob = (unsigned short*)Out;
        if (full) {
            for (int i = tid; i < 32 * 16; i += 256) {   // 16B chunks
                int r = i >> 4, c8 = i & 15;
                ((uint4*)(Ob + (size_t)(row0 + r) * 128))[c8] = ((uint4*)(sO + r * 128))[c8];
            }
        } else {
            for (int i = tid; i < 32 * 16; i += 256) {
                int r = i >> 4, c8 = i & 15;
                if (row0 + r < Nrows)
                    ((uint4*)(Ob + (size_t)(row0 + r) * 128))[c8] = ((uint4*)(sO + r * 128))[c8];
            }
        }
    } else {
        // bias -> LDS, LayerNorm + relu -> f32 out (float4 stores)
#pragma unroll
        for (int r = 0; r < 8; ++r) {
            int m = r + half * 8;
            sC[m * 128 + col]        = c0[r] + bcol;
            sC[(m + 16) * 128 + col] = c1[r] + bcol;
        }
        __syncthreads();
        // 8 threads per row, 16 cols each
        int m  = tid >> 3;       // 0..31
        int g8 = tid & 7;
        float s = 0.f, ss = 0.f;
#pragma unroll
        for (int j = 0; j < 16; ++j) {
            float v = sC[m * 128 + g8 * 16 + j];
            s += v; ss += v * v;
        }
#pragma unroll
        for (int off = 4; off >= 1; off >>= 1) {
            s  += __shfl_xor(s, off, 8);
            ss += __shfl_xor(ss, off, 8);
        }
        float mean = s * (1.0f / 128.0f);
        float var  = ss * (1.0f / 128.0f) - mean * mean;
        float rstd = rsqrtf(var + 1e-5f);
        int row = row0 + m;
        if (row < Nrows) {
            float* Of = (float*)Out;
#pragma unroll
            for (int j = 0; j < 16; j += 4) {
                float4 o;
                int cc = g8 * 16 + j;
                float v;
                v = (sC[m * 128 + cc + 0] - mean) * rstd * ln_g[cc + 0] + ln_b[cc + 0]; o.x = v > 0.f ? v : 0.f;
                v = (sC[m * 128 + cc + 1] - mean) * rstd * ln_g[cc + 1] + ln_b[cc + 1]; o.y = v > 0.f ? v : 0.f;
                v = (sC[m * 128 + cc + 2] - mean) * rstd * ln_g[cc + 2] + ln_b[cc + 2]; o.z = v > 0.f ? v : 0.f;
                v = (sC[m * 128 + cc + 3] - mean) * rstd * ln_g[cc + 3] + ln_b[cc + 3]; o.w = v > 0.f ? v : 0.f;
                ((float4*)(Of + (size_t)row * 128))[cc >> 2] = o;
            }
        }
    }
}

// ---------- gate logit: gate[i] = dot(gh[i,:], w2) + b2 ----------
__global__ void gate_dot_kernel(const unsigned short* __restrict__ gh,
                                const float* __restrict__ w2,
                                const float* __restrict__ b2,
                                float* __restrict__ gate, int N) {
    __shared__ float sw[128];
    if (threadIdx.x < 128) sw[threadIdx.x] = w2[threadIdx.x];
    __syncthreads();
    int i = blockIdx.x * blockDim.x + threadIdx.x;
    if (i < N) {
        const uint4* row = (const uint4*)(gh + (size_t)i * 128);
        float acc = 0.f;
#pragma unroll
        for (int q = 0; q < 16; ++q) {
            uint4 v = row[q];
            unsigned pk[4] = {v.x, v.y, v.z, v.w};
#pragma unroll
            for (int j = 0; j < 4; ++j) {
                unsigned p = pk[j];
                float lo = __builtin_bit_cast(float, p << 16);
                float hi = __builtin_bit_cast(float, p & 0xFFFF0000u);
                int k = q * 8 + j * 2;
                acc += lo * sw[k] + hi * sw[k + 1];
            }
        }
        gate[i] = acc + b2[0];
    }
}

__global__ void init_pool_kernel(unsigned* gmaxu, float* denom,
                                 float* pooled, int G) {
    int i = blockIdx.x * blockDim.x + threadIdx.x;
    if (i < G) { gmaxu[i] = 0u; denom[i] = 0.f; }
    if (i < G * 128) pooled[i] = 0.f;
}

__global__ void seg_max_kernel(const float* __restrict__ gate,
                               const int* __restrict__ batch,
                               unsigned* __restrict__ gmaxu, int N) {
    int i = blockIdx.x * blockDim.x + threadIdx.x;
    if (i < N) atomicMax(&gmaxu[batch[i]], enc_f(gate[i]));
}

__global__ void seg_exp_kernel(const float* __restrict__ gate,
                               const int* __restrict__ batch,
                               const unsigned* __restrict__ gmaxu,
                               float* __restrict__ ee,
                               float* __restrict__ denom, int N) {
    int i = blockIdx.x * blockDim.x + threadIdx.x;
    if (i < N) {
        int b = batch[i];
        float e = __expf(gate[i] - dec_f(gmaxu[b]));
        ee[i] = e;
        atomicAdd(&denom[b], e);
    }
}

__global__ void pool_kernel(const float* __restrict__ h1,
                            const int* __restrict__ batch,
                            const float* __restrict__ ee,
                            const float* __restrict__ denom,
                            float* __restrict__ pooled, int N) {
    int idx = blockIdx.x * blockDim.x + threadIdx.x;
    if (idx < N * 128) {
        int i = idx >> 7;
        int f = idx & 127;
        int b = batch[i];
        float alpha = ee[i] / fmaxf(denom[b], 1e-16f);
        atomicAdd(&pooled[(size_t)b * 128 + f], h1[(size_t)i * 128 + f] * alpha);
    }
}

// ---------- classifier: out[g,:2] = relu(comb·W1+b1)·W2+b2 ----------
__global__ void cls_kernel(const float* __restrict__ pooled,
                           const float* __restrict__ gf,
                           const float* __restrict__ W1,   // [160,128]
                           const float* __restrict__ b1,
                           const float* __restrict__ W2,   // [128,2]
                           const float* __restrict__ b2,
                           float* __restrict__ out, int G) {
    __shared__ float comb[160];
    __shared__ float hid[128];
    int g = blockIdx.x, t = threadIdx.x;
    for (int k = t; k < 160; k += 128)
        comb[k] = (k < 128) ? pooled[(size_t)g * 128 + k] : gf[(size_t)g * 32 + (k - 128)];
    __syncthreads();
    float acc = b1[t];
    for (int k = 0; k < 160; ++k) acc += comb[k] * W1[k * 128 + t];
    hid[t] = acc > 0.f ? acc : 0.f;
    __syncthreads();
    if (t < 2) {
        float o = b2[t];
        for (int k = 0; k < 128; ++k) o += hid[k] * W2[k * 2 + t];
        out[(size_t)g * 2 + t] = o;
    }
}

// ---------------------------------------------------------------------------
extern "C" void kernel_launch(void* const* d_in, const int* in_sizes, int n_in,
                              void* d_out, int out_size, void* d_ws, size_t ws_size,
                              hipStream_t stream) {
    const float* x     = (const float*)d_in[0];
    const int*   ei    = (const int*)d_in[1];
    const int*   batch = (const int*)d_in[3];
    const float* gf    = (const float*)d_in[4];
    const int N = in_sizes[0] / 64;
    const int E = in_sizes[1] / 2;
    const int G = in_sizes[4] / 32;

    // Locate param groups by unique sizes (robust to dict-flatten ordering).
    int iConv0 = -1, iCls = -1;
    for (int i = 5; i < n_in; ++i) {
        if (in_sizes[i] == 64 * 128)  iConv0 = i;   // conv0.lin1.W
        if (in_sizes[i] == 160 * 128) iCls = i;     // cls.lin1.W
    }
    const int iConv1 = iConv0 + 6;                  // conv1 group follows conv0
    int iGate = -1;
    for (int i = 5; i < n_in; ++i) {
        if (in_sizes[i] == 128 * 128 && i != iConv0 + 2 && i != iConv1 && i != iConv1 + 2) {
            iGate = i; break;
        }
    }
    const float* W1_0 = (const float*)d_in[iConv0 + 0];
    const float* b1_0 = (const float*)d_in[iConv0 + 1];
    const float* W2_0 = (const float*)d_in[iConv0 + 2];
    const float* b2_0 = (const float*)d_in[iConv0 + 3];
    const float* g0   = (const float*)d_in[iConv0 + 4];
    const float* t0b  = (const float*)d_in[iConv0 + 5];
    const float* W1_1 = (const float*)d_in[iConv1 + 0];
    const float* b1_1 = (const float*)d_in[iConv1 + 1];
    const float* W2_1 = (const float*)d_in[iConv1 + 2];
    const float* b2_1 = (const float*)d_in[iConv1 + 3];
    const float* g1   = (const float*)d_in[iConv1 + 4];
    const float* t1b  = (const float*)d_in[iConv1 + 5];
    const float* Wg1  = (const float*)d_in[iGate + 0];
    const float* bg1  = (const float*)d_in[iGate + 1];
    const float* wg2  = (const float*)d_in[iGate + 2];
    const float* bg2  = (const float*)d_in[iGate + 3];
    const float* Wc1  = (const float*)d_in[iCls + 0];
    const float* bc1  = (const float*)d_in[iCls + 1];
    const float* Wc2  = (const float*)d_in[iCls + 2];
    const float* bc2  = (const float*)d_in[iCls + 3];

    // Workspace carve (aliased: tbuf = t0/t1/gate-hidden, hbuf = h0/h1).
    char* w = (char*)d_ws;
    auto carve = [&](size_t bytes) -> void* {
        void* p = (void*)w;
        w += (bytes + 255) & ~(size_t)255;
        return p;
    };
    float*          z0    = (float*)carve((size_t)N * 64 * 4);
    float*          z1    = (float*)carve((size_t)N * 128 * 4);
    unsigned short* tbuf  = (unsigned short*)carve((size_t)N * 128 * 2);
    float*          hbuf  = (float*)carve((size_t)N * 128 * 4);
    float*          gatev = (float*)carve((size_t)N * 4);
    float*          ee    = (float*)carve((size_t)N * 4);
    unsigned*       gmaxu = (unsigned*)carve((size_t)G * 4);
    float*          denom = (float*)carve((size_t)G * 4);
    float*          pooled= (float*)carve((size_t)G * 128 * 4);
    unsigned short* W1_0b = (unsigned short*)carve((size_t)64 * 128 * 2);
    unsigned short* W2_0b = (unsigned short*)carve((size_t)128 * 128 * 2);
    unsigned short* W1_1b = (unsigned short*)carve((size_t)128 * 128 * 2);
    unsigned short* W2_1b = (unsigned short*)carve((size_t)128 * 128 * 2);
    unsigned short* Wg1b  = (unsigned short*)carve((size_t)128 * 128 * 2);

    auto cdiv = [](int a, int b) { return (a + b - 1) / b; };

    // 1) weights -> bf16, pre-swizzled into WMMA fragment order
    cvt_swizzle_W_kernel<64><<<cdiv(64 * 128, 256), 256, 0, stream>>>(W1_0, W1_0b);
    cvt_swizzle_W_kernel<128><<<cdiv(128 * 128, 256), 256, 0, stream>>>(W2_0, W2_0b);
    cvt_swizzle_W_kernel<128><<<cdiv(128 * 128, 256), 256, 0, stream>>>(W1_1, W1_1b);
    cvt_swizzle_W_kernel<128><<<cdiv(128 * 128, 256), 256, 0, stream>>>(W2_1, W2_1b);
    cvt_swizzle_W_kernel<128><<<cdiv(128 * 128, 256), 256, 0, stream>>>(Wg1, Wg1b);

    const int gemmBlocks = cdiv(N, 32);

    // 2) conv0:  z0 = x + scatter(x);  t0 = relu(z0 W1 + b1);  h0 = relu(LN(t0 W2 + b2))
    copy_f32_kernel<<<cdiv(N * 64, 256), 256, 0, stream>>>(x, z0, N * 64);
    scatter_add_kernel<64><<<cdiv(E * 64, 256), 256, 0, stream>>>(x, ei, E, z0);
    gemm32_wmma_kernel<64, false, 0><<<gemmBlocks, 256, 0, stream>>>(
        z0, W1_0b, b1_0, nullptr, nullptr, tbuf, N);
    gemm32_wmma_kernel<128, true, 1><<<gemmBlocks, 256, 0, stream>>>(
        tbuf, W2_0b, b2_0, g0, t0b, hbuf, N);

    // 3) conv1
    copy_f32_kernel<<<cdiv(N * 128, 256), 256, 0, stream>>>(hbuf, z1, N * 128);
    scatter_add_kernel<128><<<cdiv(E * 128, 256), 256, 0, stream>>>(hbuf, ei, E, z1);
    gemm32_wmma_kernel<128, false, 0><<<gemmBlocks, 256, 0, stream>>>(
        z1, W1_1b, b1_1, nullptr, nullptr, tbuf, N);
    gemm32_wmma_kernel<128, true, 1><<<gemmBlocks, 256, 0, stream>>>(
        tbuf, W2_1b, b2_1, g1, t1b, hbuf, N);

    // 4) gate hidden (WMMA) + logit dot
    gemm32_wmma_kernel<128, false, 0><<<gemmBlocks, 256, 0, stream>>>(
        hbuf, Wg1b, bg1, nullptr, nullptr, tbuf, N);
    gate_dot_kernel<<<cdiv(N, 256), 256, 0, stream>>>(tbuf, wg2, bg2, gatev, N);

    // 5) segment softmax + weighted pooling
    init_pool_kernel<<<cdiv(G * 128, 256), 256, 0, stream>>>(gmaxu, denom, pooled, G);
    seg_max_kernel<<<cdiv(N, 256), 256, 0, stream>>>(gatev, batch, gmaxu, N);
    seg_exp_kernel<<<cdiv(N, 256), 256, 0, stream>>>(gatev, batch, gmaxu, ee, denom, N);
    pool_kernel<<<cdiv(N * 128, 256), 256, 0, stream>>>(hbuf, batch, ee, denom, pooled, N);

    // 6) classifier -> d_out [G,2]
    cls_kernel<<<G, 128, 0, stream>>>(pooled, gf, Wc1, bc1, Wc2, bc2, (float*)d_out, G);

    (void)ws_size; (void)out_size; (void)n_in;
}
